// CVAE_UNET_Pr_17987323036522
// MI455X (gfx1250) — matrix operand order repaired and checked by reading, not compiled, not run on hardware
//
#include <hip/hip_runtime.h>
#include <math.h>

typedef __bf16 bf16_t;
typedef __attribute__((ext_vector_type(16))) __bf16 v16bf;
typedef __attribute__((ext_vector_type(8)))  float  v8f;

#define N_  2048
#define F_  128
#define T_  64
#define E_  32768
#define KP_ 1024      // pooled node count = ceil(0.5*N)
#define EPS_BN 1e-5f

// ---------------------------------------------------------------------------
// WMMA GEMM: C = EP( scale * ( (Ahi+Alo) @ (Bhi+Blo) + bias ) )
// A : M x K  row-major bf16 (hi/lo split; lo only when SPLIT)
// BT: NC x K K-major (i.e. B transposed) bf16 (hi/lo split; lo only when SPLIT)
//     -> every fragment fetch is 2 runs of 8 contiguous bf16, 16 apart = 2x global_load_b128
// one wave per 16x16 output tile; 16x16x32 bf16 WMMA, f32 accumulate.
// SPLIT is a template parameter so the 3-WMMA split-precision body is
// branch-free (no per-iteration scalar re-checks of the lo pointers).
// EP: 0 = none, 1 = relu, 2 = sigmoid
// ---------------------------------------------------------------------------
template <bool SPLIT, int EP>
__global__ void __launch_bounds__(32)
wmma_gemm_bf16(const bf16_t* __restrict__ Ahi, const bf16_t* __restrict__ Alo,
               const bf16_t* __restrict__ BThi, const bf16_t* __restrict__ BTlo,
               const float* __restrict__ bias, float* __restrict__ C,
               int M, int K, int NC, float scale)
{
    const int lane = threadIdx.x;            // 0..31
    const int i0 = blockIdx.y << 4;
    const int j0 = blockIdx.x << 4;
    const int m    = lane & 15;              // A row within tile
    const int koff = (lane >> 4) << 3;       // A-side K interleave: 0 or 8
    const int kb   = (lane >> 4) << 3;       // B-side K interleave: 0 or 8
    const int n    = j0 + (lane & 15);       // C column / BT row
    const bf16_t* arow  = Ahi + (size_t)(i0 + m) * K + koff;
    const bf16_t* alrow = SPLIT ? Alo + (size_t)(i0 + m) * K + koff : nullptr;
    const bf16_t* brow  = BThi + (size_t)n * K + kb;
    const bf16_t* blrow = SPLIT ? BTlo + (size_t)n * K + kb : nullptr;
    v8f acc = {};
    for (int k0 = 0; k0 < K; k0 += 32) {
        v16bf ah, bh;
        const bf16_t* ap = arow + k0;
#pragma unroll
        for (int e = 0; e < 8; ++e) { ah[e] = ap[e]; ah[8 + e] = ap[16 + e]; }
        const bf16_t* bp = brow + k0;
#pragma unroll
        for (int e = 0; e < 8; ++e) { bh[e] = bp[e]; bh[8 + e] = bp[16 + e]; }
        acc = __builtin_amdgcn_wmma_f32_16x16x32_bf16(false, ah, false, bh,
                                                      (short)0, acc, false, false);
        if constexpr (SPLIT) {
            // A-residual term
            v16bf al;
            const bf16_t* alp = alrow + k0;
#pragma unroll
            for (int e = 0; e < 8; ++e) { al[e] = alp[e]; al[8 + e] = alp[16 + e]; }
            acc = __builtin_amdgcn_wmma_f32_16x16x32_bf16(false, al, false, bh,
                                                          (short)0, acc, false, false);
            // B-residual term
            v16bf bl;
            const bf16_t* blp = blrow + k0;
#pragma unroll
            for (int e = 0; e < 8; ++e) { bl[e] = blp[e]; bl[8 + e] = blp[16 + e]; }
            acc = __builtin_amdgcn_wmma_f32_16x16x32_bf16(false, ah, false, bl,
                                                          (short)0, acc, false, false);
        }
    }
    const int rbase = i0 + ((lane >> 4) << 3);
    const float bv = bias ? bias[n] : 0.0f;
#pragma unroll
    for (int r = 0; r < 8; ++r) {
        float v = (acc[r] + bv) * scale;
        if constexpr (EP == 1) v = v > 0.0f ? v : 0.0f;
        if constexpr (EP == 2) v = 1.0f / (1.0f + expf(-v));
        C[(size_t)(rbase + r) * NC + n] = v;
    }
}

// ---------------------------------------------------------------------------
// small f32 GEMM (X@W), one thread per output; K <= 256 everywhere it is used
// ---------------------------------------------------------------------------
__global__ void simple_gemm(const float* __restrict__ X, const float* __restrict__ W,
                            float* __restrict__ out, int M, int K, int N)
{
    int idx = blockIdx.x * blockDim.x + threadIdx.x;
    if (idx >= M * N) return;
    int i = idx / N, j = idx - i * N;
    float s = 0.0f;
    for (int k = 0; k < K; ++k) s += X[(size_t)i * K + k] * W[(size_t)k * N + j];
    out[idx] = s;
}

// float (rows x cols, row-major) -> bf16 hi/lo split, stored TRANSPOSED (cols x rows)
__global__ void split_bf16_t(const float* __restrict__ X, bf16_t* __restrict__ hiT,
                             bf16_t* __restrict__ loT, int rows, int cols)
{
    int idx = blockIdx.x * blockDim.x + threadIdx.x;
    if (idx >= rows * cols) return;
    int k = idx / cols, n = idx - k * cols;
    float v = X[idx];
    bf16_t h = (bf16_t)v;
    hiT[(size_t)n * rows + k] = h;
    loT[(size_t)n * rows + k] = (bf16_t)(v - (float)h);
}

// adjacency scatter: A[dst, src] += 1
__global__ void build_adj(const int* __restrict__ ei, float* __restrict__ A)
{
    int e = blockIdx.x * blockDim.x + threadIdx.x;
    if (e >= E_) return;
    int src = ei[e];
    int dst = ei[E_ + e];
    atomicAdd(&A[(size_t)dst * N_ + src], 1.0f);
}

// dis[i] = rsqrt(rowsum(A)[i] + fill)   (block per row)
__global__ void rowsum_rsqrt(const float* __restrict__ A, float* __restrict__ dis,
                             int n, float fill)
{
    __shared__ float red[256];
    int i = blockIdx.x, t = threadIdx.x;
    float s = 0.0f;
    for (int j = t; j < n; j += 256) s += A[(size_t)i * n + j];
    red[t] = s; __syncthreads();
    for (int off = 128; off > 0; off >>= 1) {
        if (t < off) red[t] += red[t + off];
        __syncthreads();
    }
    if (t == 0) { float d = red[0] + fill; dis[i] = d > 0.0f ? rsqrtf(d) : 0.0f; }
}

// An = D^-1/2 (A + fill*I) D^-1/2, stored as bf16 hi/lo split (A-side only -> row-major)
__global__ void an_split(const float* __restrict__ A, const float* __restrict__ dis,
                         float fill, bf16_t* __restrict__ hi, bf16_t* __restrict__ lo, int n)
{
    int idx = blockIdx.x * blockDim.x + threadIdx.x;
    if (idx >= n * n) return;
    int i = idx / n, j = idx - i * n;
    float v = (A[idx] + ((i == j) ? fill : 0.0f)) * dis[i] * dis[j];
    bf16_t h = (bf16_t)v;
    hi[idx] = h;
    lo[idx] = (bf16_t)(v - (float)h);
}

// Aaug = A + I in bf16 (exact: small integer entries); emit row-major (A side)
// and transposed (B side) copies
__global__ void aaug_bf16(const float* __restrict__ A, bf16_t* __restrict__ out,
                          bf16_t* __restrict__ outT)
{
    int idx = blockIdx.x * blockDim.x + threadIdx.x;
    if (idx >= N_ * N_) return;
    int i = idx / N_, j = idx - i * N_;
    bf16_t v = (bf16_t)(A[idx] + ((i == j) ? 1.0f : 0.0f));
    out[idx] = v;
    outT[(size_t)j * N_ + i] = v;
}

__global__ void zero_diag(float* __restrict__ A2)
{
    int i = blockIdx.x * blockDim.x + threadIdx.x;
    if (i < N_) A2[(size_t)i * N_ + i] = 0.0f;
}

__global__ void pw_norm(const float* __restrict__ pw, float* __restrict__ out)
{
    __shared__ float red[128];
    int t = threadIdx.x;
    red[t] = pw[t] * pw[t]; __syncthreads();
    for (int off = 64; off > 0; off >>= 1) {
        if (t < off) red[t] += red[t + off];
        __syncthreads();
    }
    if (t == 0) out[0] = sqrtf(red[0]);
}

__global__ void score_kernel(const float* __restrict__ h, const float* __restrict__ pw,
                             const float* __restrict__ norm, float* __restrict__ s)
{
    int i = blockIdx.x * blockDim.x + threadIdx.x;
    if (i >= N_) return;
    float d = 0.0f;
    for (int f = 0; f < F_; ++f) d += h[(size_t)i * F_ + f] * pw[f];
    s[i] = tanhf(d / norm[0]);
}

// full in-LDS bitonic sort of 2048 (score,index) pairs, descending; perm = top 1024
__global__ void __launch_bounds__(1024)
topk_kernel(const float* __restrict__ s, int* __restrict__ perm)
{
    __shared__ float sv[2048];
    __shared__ int   si[2048];
    int t = threadIdx.x;
    for (int i = t; i < 2048; i += 1024) { sv[i] = s[i]; si[i] = i; }
    __syncthreads();
    for (int ksz = 2; ksz <= 2048; ksz <<= 1) {
        for (int j = ksz >> 1; j > 0; j >>= 1) {
            for (int i = t; i < 2048; i += 1024) {
                int l = i ^ j;
                if (l > i) {
                    bool asc = (i & ksz) != 0;           // inverted -> descending sort
                    float vi = sv[i], vl = sv[l];
                    bool sw = asc ? (vi > vl) : (vi < vl);
                    if (sw) {
                        sv[i] = vl; sv[l] = vi;
                        int ti = si[i]; si[i] = si[l]; si[l] = ti;
                    }
                }
            }
            __syncthreads();
        }
    }
    if (t < KP_) perm[t] = si[t];
}

__global__ void gather_hp(const float* __restrict__ h, const float* __restrict__ s,
                          const int* __restrict__ perm, float* __restrict__ hp)
{
    int idx = blockIdx.x * blockDim.x + threadIdx.x;
    if (idx >= KP_ * F_) return;
    int i = idx / F_, f = idx - i * F_;
    int p = perm[i];
    hp[idx] = h[(size_t)p * F_ + f] * s[p];
}

__global__ void gather_Ap(const float* __restrict__ A2, const int* __restrict__ perm,
                          float* __restrict__ Ap)
{
    int idx = blockIdx.x * blockDim.x + threadIdx.x;
    if (idx >= KP_ * KP_) return;
    int i = idx / KP_, j = idx - i * KP_;
    Ap[idx] = A2[(size_t)perm[i] * N_ + perm[j]];
}

__global__ void copy_f32(const float* __restrict__ in, float* __restrict__ out, int n)
{
    int i = blockIdx.x * blockDim.x + threadIdx.x;
    if (i < n) out[i] = in[i];
}

__global__ void scatter_add(const float* __restrict__ h2, const int* __restrict__ perm,
                            float* __restrict__ hres)
{
    int idx = blockIdx.x * blockDim.x + threadIdx.x;
    if (idx >= KP_ * F_) return;
    int i = idx / F_, f = idx - i * F_;
    hres[(size_t)perm[i] * F_ + f] += h2[idx];
}

// BatchNorm1d (batch stats, biased var); one block per feature
__global__ void batchnorm(const float* __restrict__ x, const float* __restrict__ g,
                          const float* __restrict__ b, float* __restrict__ out)
{
    __shared__ float rs[256], rq[256];
    __shared__ float sm, siv;
    int j = blockIdx.x, t = threadIdx.x;
    float s = 0.0f, q = 0.0f;
    for (int i = t; i < N_; i += 256) {
        float v = x[(size_t)i * F_ + j];
        s += v; q += v * v;
    }
    rs[t] = s; rq[t] = q; __syncthreads();
    for (int off = 128; off > 0; off >>= 1) {
        if (t < off) { rs[t] += rs[t + off]; rq[t] += rq[t + off]; }
        __syncthreads();
    }
    if (t == 0) {
        float m = rs[0] / (float)N_;
        float v = rq[0] / (float)N_ - m * m;
        sm = m; siv = rsqrtf(v + EPS_BN);
    }
    __syncthreads();
    float m = sm, iv = siv, gg = g[j], bb = b[j];
    for (int i = t; i < N_; i += 256)
        out[(size_t)i * F_ + j] = (x[(size_t)i * F_ + j] - m) * iv * gg + bb;
}

// z = [x_enc | mu + eps*exp(logstd)]
__global__ void build_z(const float* __restrict__ x_enc, const float* __restrict__ mu,
                        const float* __restrict__ logstd, const float* __restrict__ eps,
                        float* __restrict__ z)
{
    int idx = blockIdx.x * blockDim.x + threadIdx.x;
    if (idx >= N_ * F_) return;
    int i = idx / F_, f = idx - i * F_;
    z[(size_t)i * 2 * F_ + f]      = x_enc[idx];
    z[(size_t)i * 2 * F_ + F_ + f] = mu[idx] + eps[idx] * expf(logstd[idx]);
}

// ---------------------------------------------------------------------------
// host-side orchestration
// ---------------------------------------------------------------------------
struct WS {
    float *A, *A2, *Ap, *XW, *h, *hres, *hp, *h2, *x_enc, *y_post, *y_prior, *z, *tmp;
    float *dis2, *dis1, *disp, *s, *pwn;
    bf16_t *Aaug, *AaugT, *An2hi, *An2lo, *An1hi, *An1lo, *Anphi, *Anplo, *XWhi, *XWlo;
    int *perm;
};

struct UNetP { const float *w0, *b0, *pw, *w1, *b1, *wu, *bu; };

static void run_unet(hipStream_t st, const WS& w, const float* X, int cin,
                     const UNetP& p, int cout, float* out, int final_ep)
{
    // h = relu(An2 @ (X@w0) + b0)
    simple_gemm<<<(N_ * F_ + 255) / 256, 256, 0, st>>>(X, p.w0, w.XW, N_, cin, F_);
    split_bf16_t<<<(N_ * F_ + 255) / 256, 256, 0, st>>>(w.XW, w.XWhi, w.XWlo, N_, F_);
    wmma_gemm_bf16<true, 1><<<dim3(F_ / 16, N_ / 16), 32, 0, st>>>(
        w.An2hi, w.An2lo, w.XWhi, w.XWlo, p.b0, w.h, N_, N_, F_, 1.0f);
    // top-k pooling
    pw_norm<<<1, 128, 0, st>>>(p.pw, w.pwn);
    score_kernel<<<N_ / 256, 256, 0, st>>>(w.h, p.pw, w.pwn, w.s);
    topk_kernel<<<1, 1024, 0, st>>>(w.s, w.perm);
    gather_hp<<<(KP_ * F_ + 255) / 256, 256, 0, st>>>(w.h, w.s, w.perm, w.hp);
    gather_Ap<<<(KP_ * KP_ + 255) / 256, 256, 0, st>>>(w.A2, w.perm, w.Ap);
    rowsum_rsqrt<<<KP_, 256, 0, st>>>(w.Ap, w.disp, KP_, 2.0f);
    an_split<<<(KP_ * KP_ + 255) / 256, 256, 0, st>>>(w.Ap, w.disp, 2.0f, w.Anphi, w.Anplo, KP_);
    // h2 = relu(Anp @ (hp@w1) + b1)
    simple_gemm<<<(KP_ * F_ + 255) / 256, 256, 0, st>>>(w.hp, p.w1, w.XW, KP_, F_, F_);
    split_bf16_t<<<(KP_ * F_ + 255) / 256, 256, 0, st>>>(w.XW, w.XWhi, w.XWlo, KP_, F_);
    wmma_gemm_bf16<true, 1><<<dim3(F_ / 16, KP_ / 16), 32, 0, st>>>(
        w.Anphi, w.Anplo, w.XWhi, w.XWlo, p.b1, w.h2, KP_, KP_, F_, 1.0f);
    // residual unpool
    copy_f32<<<(N_ * F_ + 255) / 256, 256, 0, st>>>(w.h, w.hres, N_ * F_);
    scatter_add<<<(KP_ * F_ + 255) / 256, 256, 0, st>>>(w.h2, w.perm, w.hres);
    // out = final_ep(An2 @ (hres@wu) + bu)
    simple_gemm<<<(N_ * cout + 255) / 256, 256, 0, st>>>(w.hres, p.wu, w.XW, N_, F_, cout);
    split_bf16_t<<<(N_ * cout + 255) / 256, 256, 0, st>>>(w.XW, w.XWhi, w.XWlo, N_, cout);
    if (final_ep == 2)
        wmma_gemm_bf16<true, 2><<<dim3(cout / 16, N_ / 16), 32, 0, st>>>(
            w.An2hi, w.An2lo, w.XWhi, w.XWlo, p.bu, out, N_, N_, cout, 1.0f);
    else
        wmma_gemm_bf16<true, 0><<<dim3(cout / 16, N_ / 16), 32, 0, st>>>(
            w.An2hi, w.An2lo, w.XWhi, w.XWlo, p.bu, out, N_, N_, cout, 1.0f);
}

static void gcn_bn(hipStream_t st, const WS& w, const float* Y, const float* ww,
                   const float* bb, float scale, const float* g, const float* gb, float* out)
{
    simple_gemm<<<(N_ * F_ + 255) / 256, 256, 0, st>>>(Y, ww, w.XW, N_, F_, F_);
    split_bf16_t<<<(N_ * F_ + 255) / 256, 256, 0, st>>>(w.XW, w.XWhi, w.XWlo, N_, F_);
    wmma_gemm_bf16<true, 0><<<dim3(F_ / 16, N_ / 16), 32, 0, st>>>(
        w.An1hi, w.An1lo, w.XWhi, w.XWlo, bb, w.tmp, N_, N_, F_, scale);
    batchnorm<<<F_, 256, 0, st>>>(w.tmp, g, gb, out);
}

extern "C" void kernel_launch(void* const* d_in, const int* in_sizes, int n_in,
                              void* d_out, int out_size, void* d_ws, size_t ws_size,
                              hipStream_t stream)
{
    (void)in_sizes; (void)n_in; (void)out_size; (void)ws_size;
    int ii = 0;
    const float* x  = (const float*)d_in[ii++];
    const float* y  = (const float*)d_in[ii++];
    const int*   ei = (const int*)d_in[ii++];
    UNetP enc, post, prior, dec;
    auto grab = [&](UNetP& u) {
        u.w0 = (const float*)d_in[ii++]; u.b0 = (const float*)d_in[ii++];
        u.pw = (const float*)d_in[ii++]; u.w1 = (const float*)d_in[ii++];
        u.b1 = (const float*)d_in[ii++]; u.wu = (const float*)d_in[ii++];
        u.bu = (const float*)d_in[ii++];
    };
    grab(enc); grab(post); grab(prior); grab(dec);
    const float* mu_w    = (const float*)d_in[ii++];
    const float* mu_b    = (const float*)d_in[ii++];
    const float* sig_w   = (const float*)d_in[ii++];
    const float* sig_b   = (const float*)d_in[ii++];
    const float* mupr_w  = (const float*)d_in[ii++];
    const float* mupr_b  = (const float*)d_in[ii++];
    const float* sigpr_w = (const float*)d_in[ii++];
    const float* sigpr_b = (const float*)d_in[ii++];
    const float* bn_mu_g   = (const float*)d_in[ii++];
    const float* bn_mu_b   = (const float*)d_in[ii++];
    const float* bn_sig_g  = (const float*)d_in[ii++];
    const float* bn_sig_b  = (const float*)d_in[ii++];
    const float* bn_mupr_g = (const float*)d_in[ii++];
    const float* bn_mupr_b = (const float*)d_in[ii++];
    const float* bn_sigpr_g= (const float*)d_in[ii++];
    const float* bn_sigpr_b= (const float*)d_in[ii++];
    const float* eps_q = (const float*)d_in[ii++];
    const float* eps_p = (const float*)d_in[ii++];
    (void)eps_p;   // p_samp is not an output

    // --- workspace bump allocator (deterministic, 256B aligned) ---
    char* base = (char*)d_ws;
    size_t off = 0;
    auto alloc = [&](size_t bytes) -> char* {
        char* r = base + off;
        off = (off + bytes + 255) & ~(size_t)255;
        return r;
    };
    WS w;
    w.A     = (float*)alloc((size_t)N_ * N_ * 4);
    w.A2    = (float*)alloc((size_t)N_ * N_ * 4);
    w.Aaug  = (bf16_t*)alloc((size_t)N_ * N_ * 2);
    w.AaugT = (bf16_t*)alloc((size_t)N_ * N_ * 2);
    w.An2hi = (bf16_t*)alloc((size_t)N_ * N_ * 2);
    w.An2lo = (bf16_t*)alloc((size_t)N_ * N_ * 2);
    w.An1hi = (bf16_t*)alloc((size_t)N_ * N_ * 2);
    w.An1lo = (bf16_t*)alloc((size_t)N_ * N_ * 2);
    w.Anphi = (bf16_t*)alloc((size_t)KP_ * KP_ * 2);
    w.Anplo = (bf16_t*)alloc((size_t)KP_ * KP_ * 2);
    w.Ap    = (float*)alloc((size_t)KP_ * KP_ * 4);
    w.XW    = (float*)alloc((size_t)N_ * F_ * 4);
    w.XWhi  = (bf16_t*)alloc((size_t)N_ * F_ * 2);
    w.XWlo  = (bf16_t*)alloc((size_t)N_ * F_ * 2);
    w.h     = (float*)alloc((size_t)N_ * F_ * 4);
    w.hres  = (float*)alloc((size_t)N_ * F_ * 4);
    w.hp    = (float*)alloc((size_t)KP_ * F_ * 4);
    w.h2    = (float*)alloc((size_t)KP_ * F_ * 4);
    w.x_enc = (float*)alloc((size_t)N_ * F_ * 4);
    w.y_post= (float*)alloc((size_t)N_ * F_ * 4);
    w.y_prior=(float*)alloc((size_t)N_ * F_ * 4);
    w.z     = (float*)alloc((size_t)N_ * 2 * F_ * 4);
    w.tmp   = (float*)alloc((size_t)N_ * F_ * 4);
    w.dis2  = (float*)alloc(N_ * 4);
    w.dis1  = (float*)alloc(N_ * 4);
    w.disp  = (float*)alloc(KP_ * 4);
    w.s     = (float*)alloc(N_ * 4);
    w.pwn   = (float*)alloc(256);
    w.perm  = (int*)alloc(KP_ * 4);

    // --- output regions (concatenated tuple, return order) ---
    float* out      = (float*)d_out;
    float* o_yhat   = out;                       // N*T
    float* o_mu     = o_yhat + (size_t)N_ * T_;  // N*F
    float* o_logstd = o_mu + (size_t)N_ * F_;
    float* o_mupr   = o_logstd + (size_t)N_ * F_;
    float* o_sigpr  = o_mupr + (size_t)N_ * F_;

    // --- shared graph preprocessing ---
    hipMemsetAsync(w.A, 0, (size_t)N_ * N_ * 4, stream);
    build_adj<<<E_ / 256, 256, 0, stream>>>(ei, w.A);
    rowsum_rsqrt<<<N_, 256, 0, stream>>>(w.A, w.dis2, N_, 2.0f);
    rowsum_rsqrt<<<N_, 256, 0, stream>>>(w.A, w.dis1, N_, 1.0f);
    an_split<<<(N_ * N_ + 255) / 256, 256, 0, stream>>>(w.A, w.dis2, 2.0f, w.An2hi, w.An2lo, N_);
    an_split<<<(N_ * N_ + 255) / 256, 256, 0, stream>>>(w.A, w.dis1, 1.0f, w.An1hi, w.An1lo, N_);
    // A2 = (A+I)@(A+I) exact in bf16 (small-integer entries, f32 accumulate)
    aaug_bf16<<<(N_ * N_ + 255) / 256, 256, 0, stream>>>(w.A, w.Aaug, w.AaugT);
    wmma_gemm_bf16<false, 0><<<dim3(N_ / 16, N_ / 16), 32, 0, stream>>>(
        w.Aaug, nullptr, w.AaugT, nullptr, nullptr, w.A2, N_, N_, N_, 1.0f);
    zero_diag<<<N_ / 256, 256, 0, stream>>>(w.A2);

    // --- encoder / posterior ---
    run_unet(stream, w, x, 1, enc, F_, w.x_enc, 0);
    run_unet(stream, w, y, T_, post, F_, w.y_post, 0);
    gcn_bn(stream, w, w.y_post, mu_w,  mu_b,  1.0f, bn_mu_g,  bn_mu_b,  o_mu);
    gcn_bn(stream, w, w.y_post, sig_w, sig_b, 0.1f, bn_sig_g, bn_sig_b, o_logstd);

    // --- prior ---
    run_unet(stream, w, x, 1, prior, F_, w.y_prior, 0);
    gcn_bn(stream, w, w.y_prior, mupr_w,  mupr_b,  1.0f, bn_mupr_g,  bn_mupr_b,  o_mupr);
    gcn_bn(stream, w, w.y_prior, sigpr_w, sigpr_b, 0.1f, bn_sigpr_g, bn_sigpr_b, o_sigpr);

    // --- decoder: z = [x_enc | mu + eps_q*exp(logstd)], y_hat = sigmoid(unet(z)) ---
    build_z<<<(N_ * F_ + 255) / 256, 256, 0, stream>>>(w.x_enc, o_mu, o_logstd, eps_q, w.z);
    run_unet(stream, w, w.z, 2 * F_, dec, T_, o_yhat, 2);
}
